// SparseResBlockUpsample3d_858993459498
// MI455X (gfx1250) — matrix operand order repaired and checked by reading, not compiled, not run on hardware
//
#include <hip/hip_runtime.h>
#include <hip/hip_bf16.h>

// ---------------------------------------------------------------------------
// SparseResBlockUpsample3d for MI455X (gfx1250, wave32, WMMA bf16 16x16x32).
//
// conv1+conv2 ~43.5 GFLOP on L2-resident operands (<12MB) -> matrix-core
// bound. bf16 WMMA (K=32/instr) is 8x fp32 WMMA (K=4/instr) math density.
// Each workgroup (8 waves) computes a 64x128 output tile: every wave holds
// 4 accumulators so one B fetch (2x b128) feeds 4 WMMAs, and the next tap's
// 27-neighbor gather is prefetched into registers while the current tap's
// 32 WMMAs execute.
// ---------------------------------------------------------------------------

typedef unsigned short ushort_t;
typedef unsigned char  uchar_t;
typedef __attribute__((ext_vector_type(16))) __bf16 v16bf;
typedef __attribute__((ext_vector_type(8)))  float  v8f;

union Pack32 { uint4 q[2]; v16bf v; };

#define NVOX   2048
#define CIN    256
#define COUT   128
#define MCAND  (8 * NVOX)          // 16384
#define SGRID  66                  // (2*32 + 2) padded lookup grid
#define GRIDE  (SGRID * SGRID * SGRID)

__device__ __forceinline__ ushort_t f2bf(float f) {
  unsigned u = __float_as_uint(f);
  u += 0x7FFFu + ((u >> 16) & 1u);           // round-to-nearest-even
  return (ushort_t)(u >> 16);
}

// ---------------------------------------------------------------- prep ----

__global__ void k_subdiv(const float* __restrict__ feats,
                         const float* __restrict__ sw,
                         const float* __restrict__ sb,
                         float* __restrict__ subdiv_ws,
                         float* __restrict__ out_tail) {
  int t = blockIdx.x * blockDim.x + threadIdx.x;      // N*8 threads
  if (t >= NVOX * 8) return;
  int n = t >> 3, j = t & 7;
  const float* f = feats + (size_t)n * CIN;
  float s = sb[j];
  for (int c = 0; c < CIN; ++c) s += f[c] * sw[c * 8 + j];
  subdiv_ws[t] = s;
  out_tail[t]  = s;
}

__global__ void k_lnsilu(const float* __restrict__ feats,
                         const float* __restrict__ w,
                         const float* __restrict__ b,
                         ushort_t* __restrict__ hbf) {
  int n = blockIdx.x, c = threadIdx.x;                // 256 threads
  __shared__ float sS[CIN], sQ[CIN];
  float x = feats[(size_t)n * CIN + c];
  sS[c] = x; sQ[c] = x * x;
  __syncthreads();
  for (int o = CIN / 2; o > 0; o >>= 1) {
    if (c < o) { sS[c] += sS[c + o]; sQ[c] += sQ[c + o]; }
    __syncthreads();
  }
  float mean = sS[0] * (1.f / CIN);
  float var  = sQ[0] * (1.f / CIN) - mean * mean;
  float y = (x - mean) * rsqrtf(var + 1e-6f) * w[c] + b[c];
  float h = y / (1.f + __expf(-y));
  hbf[(size_t)n * CIN + c] = f2bf(h);
}

__global__ void k_skip(const float* __restrict__ feats,
                       const float* __restrict__ w,
                       const float* __restrict__ b,
                       float* __restrict__ sk) {
  int t = blockIdx.x * blockDim.x + threadIdx.x;      // N*128 threads
  if (t >= NVOX * COUT) return;
  int n = t >> 7, co = t & 127;
  const float* f = feats + (size_t)n * CIN;
  float s = b[co];
  for (int c = 0; c < CIN; ++c) s += f[c] * w[c * COUT + co];
  sk[t] = s;
}

// repack conv1_w [27][256][128] fp32 -> bf16 B-layout [k][c8][g][n128][i16]
__global__ void k_cw1(const float* __restrict__ w, ushort_t* __restrict__ wp) {
  int e = blockIdx.x * blockDim.x + threadIdx.x;
  if (e >= 27 * 8 * 2 * 128 * 16) return;
  int i = e & 15, n = (e >> 4) & 127, g = (e >> 11) & 1,
      c = (e >> 12) & 7, k = e >> 15;
  int ch = c * 32 + g * 8 + (i & 7) + (i >> 3) * 16;
  wp[e] = f2bf(w[((size_t)k * CIN + ch) * COUT + n]);
}

// repack conv2_w [27][128][128] -> [k][c4][g][n128][i16]
__global__ void k_cw2(const float* __restrict__ w, ushort_t* __restrict__ wp) {
  int e = blockIdx.x * blockDim.x + threadIdx.x;
  if (e >= 27 * 4 * 2 * 128 * 16) return;
  int i = e & 15, n = (e >> 4) & 127, g = (e >> 11) & 1,
      c = (e >> 12) & 3, k = e >> 14;
  int ch = c * 32 + g * 8 + (i & 7) + (i >> 3) * 16;
  wp[e] = f2bf(w[((size_t)k * COUT + ch) * COUT + n]);
}

// ------------------------------------------------------------ topology ----

__global__ void k_gridfill(int* __restrict__ g) {
  int t = blockIdx.x * blockDim.x + threadIdx.x;
  if (t < GRIDE) g[t] = MCAND;                        // sentinel -> zero row
}

__global__ void k_scatter(const int* __restrict__ coords, int* __restrict__ g) {
  int m = blockIdx.x * blockDim.x + threadIdx.x;
  if (m >= MCAND) return;
  int p = m >> 3, j = m & 7;
  int x = coords[p * 3 + 0] * 2 + ((j >> 2) & 1);
  int y = coords[p * 3 + 1] * 2 + ((j >> 1) & 1);
  int z = coords[p * 3 + 2] * 2 + (j & 1);
  g[(x + 1) * SGRID * SGRID + (y + 1) * SGRID + (z + 1)] = m;
}

__global__ void k_nidx(const int* __restrict__ coords,
                       const int* __restrict__ g, int* __restrict__ nidx) {
  int t = blockIdx.x * blockDim.x + threadIdx.x;
  if (t >= MCAND * 27) return;
  int m = t / 27, tap = t % 27;
  int p = m >> 3, j = m & 7;
  int x = coords[p * 3 + 0] * 2 + ((j >> 2) & 1) + 1 + (tap / 9 - 1);
  int y = coords[p * 3 + 1] * 2 + ((j >> 1) & 1) + 1 + ((tap / 3) % 3 - 1);
  int z = coords[p * 3 + 2] * 2 + (j & 1) + 1 + (tap % 3 - 1);
  nidx[t] = g[x * SGRID * SGRID + y * SGRID + z];
}

__global__ void k_children(const ushort_t* __restrict__ hbf,
                           const float* __restrict__ subdiv,
                           ushort_t* __restrict__ hc,
                           uchar_t* __restrict__ mask,
                           ushort_t* __restrict__ h1) {
  int m = blockIdx.x, c = threadIdx.x;                // M+1 blocks x 256
  if (m == MCAND) {
    hc[(size_t)MCAND * CIN + c] = 0;
    if (c < COUT) h1[(size_t)MCAND * COUT + c] = 0;
    return;
  }
  bool msk = subdiv[m] > 0.f;
  hc[(size_t)m * CIN + c] = msk ? hbf[(size_t)(m >> 3) * CIN + c] : (ushort_t)0;
  if (c == 0) mask[m] = msk ? 1 : 0;
}

// ---------------------------------------------------------- WMMA convs ----

// conv1: 64x128 tile/WG, 4 acc per wave; LN (no affine) + SiLU -> bf16 h1
__global__ void __launch_bounds__(256)
k_conv1(const ushort_t* __restrict__ hc, const int* __restrict__ nidx,
        const ushort_t* __restrict__ w1p, const float* __restrict__ bias,
        ushort_t* __restrict__ h1) {
  const int tid  = threadIdx.x;
  const int wave = tid >> 5, lane = tid & 31;
  const int g = lane >> 4, nl = lane & 15;
  const int m0 = blockIdx.x * 64;

  __shared__ __align__(16) ushort_t sA[64 * CIN];     // 32 KB A tile
  __shared__ int   sIdx[27 * 64];
  __shared__ float sRedS[256], sRedQ[256];
  __shared__ float sMean[64], sInv[64];

  for (int i = tid; i < 27 * 64; i += 256) {
    int k = i >> 6, r = i & 63;
    sIdx[i] = nidx[(size_t)(m0 + r) * 27 + k];
  }
  __syncthreads();

  // register-prefetched gather: 64 rows x 512B / 256 thr = 8 x 16B per thread
  uint4 st[8];
#pragma unroll
  for (int j = 0; j < 8; ++j) {
    int sid = tid + 256 * j, row = sid >> 5, s = sid & 31;
    st[j] = *reinterpret_cast<const uint4*>(
        hc + (size_t)sIdx[row] * CIN + s * 8);
  }

  v8f acc[4] = {v8f{}, v8f{}, v8f{}, v8f{}};
  for (int k = 0; k < 27; ++k) {
#pragma unroll
    for (int j = 0; j < 8; ++j) {                     // store tap k to LDS
      int sid = tid + 256 * j, row = sid >> 5, s = sid & 31;
      *reinterpret_cast<uint4*>(sA + row * CIN + s * 8) = st[j];
    }
    __syncthreads();
    if (k + 1 < 27) {                                 // issue tap k+1 loads now
#pragma unroll
      for (int j = 0; j < 8; ++j) {
        int sid = tid + 256 * j, row = sid >> 5, s = sid & 31;
        st[j] = *reinterpret_cast<const uint4*>(
            hc + (size_t)sIdx[(k + 1) * 64 + row] * CIN + s * 8);
      }
    }
#pragma unroll
    for (int c = 0; c < 8; ++c) {
      Pack32 pb;
      const uint4* bp = reinterpret_cast<const uint4*>(
          w1p + (size_t)((((k * 8 + c) * 2 + g) * 128) + (wave * 16 + nl)) * 16);
      pb.q[0] = bp[0]; pb.q[1] = bp[1];
#pragma unroll
      for (int t = 0; t < 4; ++t) {                   // 4 M-tiles reuse B
        Pack32 pa;
        const uint4* ap = reinterpret_cast<const uint4*>(
            sA + (t * 16 + nl) * CIN + c * 32 + g * 8);
        pa.q[0] = ap[0];                              // K = c*32 + g*8 + [0,8)
        pa.q[1] = ap[2];                              // K = c*32+16+g*8 + [0,8)
        acc[t] = __builtin_amdgcn_wmma_f32_16x16x32_bf16(
            false, pa.v, false, pb.v, (short)0, acc[t], false, false);
      }
    }
    __syncthreads();                                  // done reading sA
  }

  // epilogue: bias + LayerNorm(no affine) + SiLU; fp32 tile aliases sA
  float* sC = reinterpret_cast<float*>(sA);           // 64*128*4 = 32 KB
#pragma unroll
  for (int t = 0; t < 4; ++t)
#pragma unroll
    for (int v = 0; v < 8; ++v) {
      int row = t * 16 + v + 8 * g, col = wave * 16 + nl;
      sC[row * COUT + col] = acc[t][v] + bias[col];
    }
  __syncthreads();
  {
    int row = tid >> 2, sub = tid & 3;                // 4 threads per row
    float s = 0.f, q = 0.f;
#pragma unroll
    for (int i = 0; i < 32; ++i) {
      float x = sC[row * COUT + sub * 32 + i];
      s += x; q += x * x;
    }
    sRedS[tid] = s; sRedQ[tid] = q;
  }
  __syncthreads();
  if (tid < 64) {
    float s = 0.f, q = 0.f;
#pragma unroll
    for (int i = 0; i < 4; ++i) { s += sRedS[tid * 4 + i]; q += sRedQ[tid * 4 + i]; }
    float mean = s * (1.f / COUT);
    float var  = q * (1.f / COUT) - mean * mean;
    sMean[tid] = mean; sInv[tid] = rsqrtf(var + 1e-6f);
  }
  __syncthreads();
  {
    int row = tid >> 2, sub = tid & 3;
    float mean = sMean[row], inv = sInv[row];
#pragma unroll
    for (int i = 0; i < 32; ++i) {
      int col = sub * 32 + i;
      float y = (sC[row * COUT + col] - mean) * inv;
      float h = y / (1.f + __expf(-y));
      h1[(size_t)(m0 + row) * COUT + col] = f2bf(h);
    }
  }
}

// conv2: 64x128 tile/WG + bias + parent skip + mask -> fp32 out
__global__ void __launch_bounds__(256)
k_conv2(const ushort_t* __restrict__ h1, const int* __restrict__ nidx,
        const ushort_t* __restrict__ w2p, const float* __restrict__ bias,
        const float* __restrict__ sk, const uchar_t* __restrict__ mask,
        float* __restrict__ out) {
  const int tid  = threadIdx.x;
  const int wave = tid >> 5, lane = tid & 31;
  const int g = lane >> 4, nl = lane & 15;
  const int m0 = blockIdx.x * 64;

  __shared__ __align__(16) ushort_t sA[64 * COUT];    // 16 KB
  __shared__ int sIdx[27 * 64];

  for (int i = tid; i < 27 * 64; i += 256) {
    int k = i >> 6, r = i & 63;
    sIdx[i] = nidx[(size_t)(m0 + r) * 27 + k];
  }
  __syncthreads();

  uint4 st[4];                                        // 64 rows x 256B / 256 thr
#pragma unroll
  for (int j = 0; j < 4; ++j) {
    int sid = tid + 256 * j, row = sid >> 4, s = sid & 15;
    st[j] = *reinterpret_cast<const uint4*>(
        h1 + (size_t)sIdx[row] * COUT + s * 8);
  }

  v8f acc[4] = {v8f{}, v8f{}, v8f{}, v8f{}};
  for (int k = 0; k < 27; ++k) {
#pragma unroll
    for (int j = 0; j < 4; ++j) {
      int sid = tid + 256 * j, row = sid >> 4, s = sid & 15;
      *reinterpret_cast<uint4*>(sA + row * COUT + s * 8) = st[j];
    }
    __syncthreads();
    if (k + 1 < 27) {
#pragma unroll
      for (int j = 0; j < 4; ++j) {
        int sid = tid + 256 * j, row = sid >> 4, s = sid & 15;
        st[j] = *reinterpret_cast<const uint4*>(
            h1 + (size_t)sIdx[(k + 1) * 64 + row] * COUT + s * 8);
      }
    }
#pragma unroll
    for (int c = 0; c < 4; ++c) {
      Pack32 pb;
      const uint4* bp = reinterpret_cast<const uint4*>(
          w2p + (size_t)((((k * 4 + c) * 2 + g) * 128) + (wave * 16 + nl)) * 16);
      pb.q[0] = bp[0]; pb.q[1] = bp[1];
#pragma unroll
      for (int t = 0; t < 4; ++t) {
        Pack32 pa;
        const uint4* ap = reinterpret_cast<const uint4*>(
            sA + (t * 16 + nl) * COUT + c * 32 + g * 8);
        pa.q[0] = ap[0];
        pa.q[1] = ap[2];
        acc[t] = __builtin_amdgcn_wmma_f32_16x16x32_bf16(
            false, pa.v, false, pb.v, (short)0, acc[t], false, false);
      }
    }
    __syncthreads();
  }

#pragma unroll
  for (int t = 0; t < 4; ++t)
#pragma unroll
    for (int v = 0; v < 8; ++v) {
      int row = t * 16 + v + 8 * g, col = wave * 16 + nl;
      int m = m0 + row;
      float maskf = mask[m] ? 1.f : 0.f;
      out[(size_t)m * COUT + col] =
          (acc[t][v] + bias[col] + sk[(size_t)(m >> 3) * COUT + col]) * maskf;
    }
}

// ---------------------------------------------------------------- host ----

extern "C" void kernel_launch(void* const* d_in, const int* in_sizes, int n_in,
                              void* d_out, int out_size, void* d_ws, size_t ws_size,
                              hipStream_t stream) {
  const float* feats    = (const float*)d_in[0];
  const float* norm1_w  = (const float*)d_in[1];
  const float* norm1_b  = (const float*)d_in[2];
  const float* subdiv_w = (const float*)d_in[3];
  const float* subdiv_b = (const float*)d_in[4];
  const float* conv1_w  = (const float*)d_in[5];
  const float* conv1_b  = (const float*)d_in[6];
  const float* conv2_w  = (const float*)d_in[7];
  const float* conv2_b  = (const float*)d_in[8];
  const float* skip_w   = (const float*)d_in[9];
  const float* skip_b   = (const float*)d_in[10];
  const int*   coords   = (const int*)d_in[11];

  float* out      = (float*)d_out;                    // [M,128]
  float* out_tail = out + (size_t)MCAND * COUT;       // [N,8] subdiv

  char* ws = (char*)d_ws;                             // ~20.4 MB total
  size_t off = 0;
  auto take = [&](size_t bytes) {
    void* p = ws + off;
    off = (off + bytes + 511) & ~(size_t)511;
    return p;
  };
  int*      ggrid = (int*)take((size_t)GRIDE * 4);
  int*      nidx  = (int*)take((size_t)MCAND * 27 * 4);
  float*    sdws  = (float*)take((size_t)NVOX * 8 * 4);
  uchar_t*  mask  = (uchar_t*)take((size_t)MCAND);
  ushort_t* hbf   = (ushort_t*)take((size_t)NVOX * CIN * 2);
  ushort_t* hc    = (ushort_t*)take((size_t)(MCAND + 1) * CIN * 2);
  ushort_t* h1    = (ushort_t*)take((size_t)(MCAND + 1) * COUT * 2);
  float*    sk    = (float*)take((size_t)NVOX * COUT * 4);
  ushort_t* w1p   = (ushort_t*)take((size_t)27 * 8 * 2 * 128 * 16 * 2);
  ushort_t* w2p   = (ushort_t*)take((size_t)27 * 4 * 2 * 128 * 16 * 2);

  k_cw1<<<(27 * 8 * 2 * 128 * 16 + 255) / 256, 256, 0, stream>>>(conv1_w, w1p);
  k_cw2<<<(27 * 4 * 2 * 128 * 16 + 255) / 256, 256, 0, stream>>>(conv2_w, w2p);
  k_subdiv<<<(NVOX * 8 + 255) / 256, 256, 0, stream>>>(feats, subdiv_w, subdiv_b, sdws, out_tail);
  k_lnsilu<<<NVOX, 256, 0, stream>>>(feats, norm1_w, norm1_b, hbf);
  k_skip<<<(NVOX * COUT + 255) / 256, 256, 0, stream>>>(feats, skip_w, skip_b, sk);

  k_gridfill<<<(GRIDE + 255) / 256, 256, 0, stream>>>(ggrid);
  k_scatter<<<(MCAND + 255) / 256, 256, 0, stream>>>(coords, ggrid);
  k_children<<<MCAND + 1, 256, 0, stream>>>(hbf, sdws, hc, mask, h1);
  k_nidx<<<(MCAND * 27 + 255) / 256, 256, 0, stream>>>(coords, ggrid, nidx);

  k_conv1<<<MCAND / 64, 256, 0, stream>>>(hc, nidx, w1p, conv1_b, h1);
  k_conv2<<<MCAND / 64, 256, 0, stream>>>(h1, nidx, w2p, conv2_b, sk, mask, out);
}
